// QLSTM_1bit_44633300140400
// MI455X (gfx1250) — 8-bit matrix operand order repaired and checked by reading, compiled, not run on hardware
//
#include <hip/hip_runtime.h>
#include <hip/hip_bf16.h>


typedef __attribute__((ext_vector_type(8))) int v8i;

#define Bq   32
#define Sq   512
#define Vq   32000
#define Eq   512
#define Hq   1024
#define Oq   2
#define GCOFF 2048

#define WS_MAX   0
#define WS_XA    1024
#define WS_XA_SZ (8192u*1024u)
#define WS_WIH   (WS_XA + WS_XA_SZ)
#define WS_WIH_SZ (512u*1024u)
#define WS_WHH   (WS_WIH + WS_WIH_SZ)
#define WS_WHH_SZ (1024u*1024u)
#define WS_XGB   (WS_WHH + WS_WHH_SZ)
#define WS_XGB_SZ (16384u*1024u*2u)
#define WS_HFIN  (WS_XGB + WS_XGB_SZ)

__device__ __forceinline__ int a_kstart(int v, int half) {
    int vv = v & 3, g = v >> 2;
    return g * 32 + ((vv >> 1) << 4) + ((vv & 1) << 2) + (half << 3);
}
__device__ __forceinline__ int b_kstart(int v, int half) {
    return ((v >> 2) << 5) + (((v & 3) >> 1) << 4) + ((v & 1) << 2) + (half << 3);
}

__global__ void init_max_kernel(unsigned* p) {
    if (threadIdx.x == 0 && blockIdx.x == 0) *p = 0u;
}

__global__ __launch_bounds__(256) void maxabs_kernel(const float* __restrict__ w,
                                                     int n, unsigned* __restrict__ out) {
    __shared__ unsigned red[256];
    unsigned m = 0;
    for (int i = blockIdx.x * 256 + threadIdx.x; i < n; i += gridDim.x * 256)
        m = max(m, __float_as_uint(fabsf(w[i])));
    red[threadIdx.x] = m;
    __syncthreads();
    for (int off = 128; off > 0; off >>= 1) {
        if (threadIdx.x < off)
            red[threadIdx.x] = max(red[threadIdx.x], red[threadIdx.x + off]);
        __syncthreads();
    }
    if (threadIdx.x == 0) atomicMax(out, red[0]);
}

__global__ __launch_bounds__(32) void pack_x_kernel(const int* __restrict__ text,
                                                    const float* __restrict__ emb,
                                                    const unsigned* __restrict__ maxbits,
                                                    signed char* __restrict__ xA) {
    const int tile = blockIdx.x;
    const int tm = tile >> 3, tk = tile & 7;
    const int lane = threadIdx.x;
    const int mloc = lane & 15, half = lane >> 4;
    const int m = tm * 16 + mloc;
    const int s = m >> 5, b = m & 31;
    const int row = text[b * Sq + s];
    const float inv = 127.0f / __uint_as_float(*maxbits);
    signed char* dst = xA + (size_t)tile * 1024 + lane * 32;
    const float* src = emb + (size_t)row * Eq + tk * 64;
#pragma unroll
    for (int v = 0; v < 8; ++v) {
        const int ks = a_kstart(v, half);
#pragma unroll
        for (int j = 0; j < 4; ++j) {
            float q = rintf(src[ks + j] * inv);
            dst[v * 4 + j] = (q >= 0.0f) ? (signed char)1 : (signed char)-1;
        }
    }
}

__global__ __launch_bounds__(32) void pack_wB_kernel(const float* __restrict__ W,
                                                     signed char* __restrict__ dst,
                                                     int K, int Ntiles) {
    const int tile = blockIdx.x;
    const int tk = tile / Ntiles, tn = tile % Ntiles;
    const int lane = threadIdx.x;
    const int nloc = lane & 15, half = lane >> 4;
    const int n = tn * 16 + nloc;
    signed char* d = dst + (size_t)tile * 1024 + lane * 32;
    const float* src = W + (size_t)(GCOFF + n) * K + tk * 64;
#pragma unroll
    for (int v = 0; v < 8; ++v) {
        const int ks = b_kstart(v, half);
#pragma unroll
        for (int j = 0; j < 4; ++j)
            d[v * 4 + j] = (src[ks + j] >= 0.0f) ? (signed char)1 : (signed char)-1;
    }
}

__global__ __launch_bounds__(256) void xg_gemm_kernel(const signed char* __restrict__ xA,
                                                      const signed char* __restrict__ wB,
                                                      short* __restrict__ xgb) {
    const int wtile = blockIdx.x * 8 + (threadIdx.x >> 5);
    const int lane = threadIdx.x & 31;
    const int tileM = wtile >> 6;
    const int tileN = wtile & 63;
    v8i acc = (v8i)0;
#pragma unroll
    for (int kt = 0; kt < 8; ++kt) {
        v8i a = *(const v8i*)(xA + ((size_t)(tileM * 8 + kt)) * 1024 + lane * 32);
        v8i b = *(const v8i*)(wB + ((size_t)(kt * 64 + tileN)) * 1024 + lane * 32);
        acc = __builtin_amdgcn_wmma_i32_16x16x64_iu8(true, a, true, b, acc, false, false);
    }
    const int n = lane & 15, half = lane >> 4;
    const int j = tileN * 16 + n;
#pragma unroll
    for (int r = 0; r < 8; ++r) {
        const int m = tileM * 16 + half * 8 + r;
        xgb[(size_t)m * 1024 + j] = (short)acc[r];
    }
}

__global__ __launch_bounds__(1024) void scan_kernel(const short* __restrict__ xgb,
                                                    const signed char* __restrict__ whhF,
                                                    const float* __restrict__ b_ih,
                                                    const float* __restrict__ b_hh,
                                                    signed char* __restrict__ hS_out) {
    __shared__ signed char hS[Bq * Hq];
    __shared__ float biasS[Hq];

    const int tid = threadIdx.x;
    const int wave = tid >> 5, lane = tid & 31;
    const int n = lane & 15, half = lane >> 4;

    for (int i = tid; i < Bq * Hq; i += 1024) hS[i] = 1;
    if (tid < Hq) biasS[tid] = b_ih[GCOFF + tid] + b_hh[GCOFF + tid];
    __syncthreads();

    int c[4][8];
#pragma unroll
    for (int t = 0; t < 4; ++t)
#pragma unroll
        for (int r = 0; r < 8; ++r) c[t][r] = 0;

    for (int s = 0; s < Sq; ++s) {
        if (s + 1 < Sq) {
            const char* nx = (const char*)(xgb + (size_t)(s + 1) * (Bq * Hq));
            __builtin_prefetch(nx + tid * 64, 0, 1);
        }

        signed char hnew[4][8];
#pragma unroll
        for (int t = 0; t < 4; ++t) {
            const int tile = wave * 4 + t;
            const int mt = tile >> 6;
            const int nt = tile & 63;
            v8i acc = (v8i)0;
            const int mrow = mt * 16 + n;
            for (int kt = 0; kt < 16; ++kt) {
                v8i a;
#pragma unroll
                for (int v = 0; v < 8; ++v)
                    a[v] = *(const int*)&hS[mrow * Hq + kt * 64 + a_kstart(v, half)];
                v8i b = *(const v8i*)(whhF + ((size_t)(kt * 64 + nt)) * 1024 + lane * 32);
                acc = __builtin_amdgcn_wmma_i32_16x16x64_iu8(true, a, true, b, acc,
                                                             false, false);
            }
            const int j = nt * 16 + n;
            const float bias = biasS[j];
#pragma unroll
            for (int r = 0; r < 8; ++r) {
                const int bb = mt * 16 + half * 8 + r;
                const float pre = (float)acc[r] +
                                  (float)xgb[((size_t)s * Bq + bb) * Hq + j] + bias;
                const int gc = (pre >= 0.0f) ? 1 : -1;
                const int cn = c[t][r] + gc;
                c[t][r] = cn;
                hnew[t][r] = (cn >= 0) ? (signed char)1 : (signed char)-1;
            }
        }
        __syncthreads();
#pragma unroll
        for (int t = 0; t < 4; ++t) {
            const int tile = wave * 4 + t;
            const int mt = tile >> 6, nt = tile & 63;
            const int j = nt * 16 + n;
#pragma unroll
            for (int r = 0; r < 8; ++r)
                hS[(mt * 16 + half * 8 + r) * Hq + j] = hnew[t][r];
        }
        __syncthreads();
    }

    for (int i = tid; i < Bq * Hq; i += 1024) hS_out[i] = hS[i];
}

__global__ __launch_bounds__(64) void fc_kernel(const signed char* __restrict__ hS,
                                                const float* __restrict__ fcW,
                                                float* __restrict__ out) {
    const int t = threadIdx.x;
    if (t >= Bq * Oq) return;
    const int b = t >> 1, o = t & 1;
    int acc = 0;
    for (int k = 0; k < Hq; ++k) {
        const int hv = hS[b * Hq + k];
        const int w = (fcW[o * Hq + k] >= 0.0f) ? 1 : -1;
        acc += hv * w;
    }
    out[b * Oq + o] = (float)acc;
}

extern "C" void kernel_launch(void* const* d_in, const int* in_sizes, int n_in,
                              void* d_out, int out_size, void* d_ws, size_t ws_size,
                              hipStream_t stream) {
    const int*   text   = (const int*)d_in[0];
    const float* emb_W  = (const float*)d_in[2];
    const float* W_ih   = (const float*)d_in[3];
    const float* W_hh   = (const float*)d_in[4];
    const float* b_ih   = (const float*)d_in[5];
    const float* b_hh   = (const float*)d_in[6];
    const float* fc_W   = (const float*)d_in[7];

    char* ws = (char*)d_ws;
    unsigned*     maxp = (unsigned*)(ws + WS_MAX);
    signed char*  xA   = (signed char*)(ws + WS_XA);
    signed char*  wihB = (signed char*)(ws + WS_WIH);
    signed char*  whhB = (signed char*)(ws + WS_WHH);
    short*        xgb  = (short*)(ws + WS_XGB);
    signed char*  hfin = (signed char*)(ws + WS_HFIN);

    init_max_kernel<<<1, 32, 0, stream>>>(maxp);
    maxabs_kernel<<<1024, 256, 0, stream>>>(emb_W, Vq * Eq, maxp);
    pack_x_kernel<<<8192, 32, 0, stream>>>(text, emb_W, maxp, xA);
    pack_wB_kernel<<<8 * 64, 32, 0, stream>>>(W_ih, wihB, Eq, 64);
    pack_wB_kernel<<<16 * 64, 32, 0, stream>>>(W_hh, whhB, Hq, 64);
    xg_gemm_kernel<<<8192, 256, 0, stream>>>(xA, wihB, xgb);
    scan_kernel<<<1, 1024, 0, stream>>>(xgb, whhB, b_ih, b_hh, hfin);
    fc_kernel<<<1, 64, 0, stream>>>(hfin, fc_W, (float*)d_out);
}
